// IterativeTTTLayerV2_62826781606574
// MI455X (gfx1250) — matrix operand order repaired and checked by reading, not compiled, not run on hardware
//
#include <hip/hip_runtime.h>

typedef __attribute__((ext_vector_type(16))) __bf16 v16bf;
typedef __attribute__((ext_vector_type(8)))  float  v8f;
typedef __attribute__((ext_vector_type(2)))  float  v2f;
typedef __attribute__((ext_vector_type(4)))  unsigned int vu4;
typedef __attribute__((ext_vector_type(8)))  int    vi8;
typedef __attribute__((ext_vector_type(4)))  int    vi4;

#define TOK_M   8192     // B * S
#define EMB     1024
#define HEADS   16
#define DHEAD   64
#define SEQ     2048
#define NBATCH  4
#define LRC     0.1f
#define EPSC    1e-5f

// ---------------------------------------------------------------------------
// Utility: wave32 all-lane sum
// ---------------------------------------------------------------------------
__device__ __forceinline__ float wave_sum(float v) {
#pragma unroll
  for (int off = 16; off > 0; off >>= 1) v += __shfl_xor(v, off, 32);
  return v;
}

// ---------------------------------------------------------------------------
// f32 -> bf16 cast
// ---------------------------------------------------------------------------
__global__ void cast_f32_to_bf16(const float* __restrict__ in,
                                 __bf16* __restrict__ out, int n) {
  for (int i = blockIdx.x * blockDim.x + threadIdx.x; i < n;
       i += gridDim.x * blockDim.x)
    out[i] = (__bf16)in[i];
}

// ---------------------------------------------------------------------------
// bf16 NT GEMM: C[m,n] = sum_k A[m,k] * B[n,k]   (torch Linear: X @ W^T)
// Block: 256 threads (8 waves), tile 128x128, K-step 32, double-buffered LDS.
// Tile loads via Tensor Data Mover (TDM) with hardware LDS padding that
// reproduces the LDT=40 layout; wave 0 issues, TENSORcnt + barrier sync.
// MODE 0: C row-major [M,N] f32 ; MODE 1: scatter to [H,S,B,D] f32
// ---------------------------------------------------------------------------
#define BM 128
#define BN 128
#define BK 32
#define LDT 40   // bf16 elements per LDS tile row (32 + pad 8)

#if __has_builtin(__builtin_amdgcn_tensor_load_to_lds) && \
    __has_builtin(__builtin_amdgcn_s_wait_tensorcnt)
#define HAS_TDM 1
#else
#define HAS_TDM 0
#endif

#if HAS_TDM
// D# groups per CDNA5 ISA: 2-D tensor, data_size=2B,
// tile 32(K) x 128(rows), pad 4 dwords after every 16 dwords (-> LDS stride 40).
// clang-23 toolchain: 6-arg builtin (g0 v4u, g1 v8i, g2 v4i, g3 v4i, v8i, cpol).
__device__ __forceinline__ void tdm_load_tile(const __bf16* gsrc,
                                              unsigned lds_off,
                                              int tensor_rows, int row_stride) {
  const unsigned long long ga = (unsigned long long)gsrc;
  vu4 g0;
  g0[0] = 1u;                                            // count=1, no gather
  g0[1] = lds_off;                                       // LDS byte address
  g0[2] = (unsigned)ga;                                  // global_addr lo
  g0[3] = (unsigned)((ga >> 32) & 0x1ffffffull) | (2u << 30);  // hi25 | type=2
  vi8 g1;
  g1[0] = (1 << 16)        // data_size = 2 bytes
        | (1 << 20)        // pad_enable
        | (3 << 22)        // pad_interval: 16 dwords
        | (3 << 25);       // pad_amount: 4 dwords
  g1[1] = (int)(((unsigned)row_stride & 0xffffu) << 16);          // dim0 lo16
  g1[2] = (int)((((unsigned)row_stride >> 16) & 0xffffu) |
                (((unsigned)tensor_rows & 0xffffu) << 16));       // dim0 hi | dim1 lo
  g1[3] = (int)(((unsigned)tensor_rows >> 16) & 0xffffu) | (32 << 16); // dim1 hi | tile_dim0=32
  g1[4] = BM;              // tile_dim1 = 128 rows; tile_dim2 = 0
  g1[5] = row_stride;      // tensor_dim0_stride (elements), lo32
  g1[6] = 0;
  g1[7] = 0;
  const vi4 gz4 = {0, 0, 0, 0};
  const vi8 gz8 = {0, 0, 0, 0, 0, 0, 0, 0};
  __builtin_amdgcn_tensor_load_to_lds(g0, g1, gz4, gz4, gz8, 0);
}
#endif

template <int MODE>
__global__ __launch_bounds__(256)
void gemm_bf16_nt(const __bf16* __restrict__ A, const __bf16* __restrict__ B,
                  float* __restrict__ C, int M, int N, int K)
{
  __shared__ __bf16 As[2][BM * LDT];
  __shared__ __bf16 Bs[2][BN * LDT];

  const int tid  = threadIdx.x;
  const int lane = tid & 31;
  const int wid  = tid >> 5;
  const int wm   = wid & 3;        // 0..3 -> 32 rows each
  const int wn   = wid >> 2;       // 0..1 -> 64 cols each
  const int m0   = blockIdx.y * BM;
  const int n0   = blockIdx.x * BN;

  v8f acc[2][4];
#pragma unroll
  for (int i = 0; i < 2; ++i)
#pragma unroll
    for (int j = 0; j < 4; ++j)
#pragma unroll
      for (int r = 0; r < 8; ++r) acc[i][j][r] = 0.0f;

  const int fm = lane & 15;          // fragment row/col
  const int kh = lane >> 4;          // K-half selector
  const int nsteps = K / BK;

#if HAS_TDM
  const unsigned as_off[2] = {(unsigned)(unsigned long long)&As[0][0],
                              (unsigned)(unsigned long long)&As[1][0]};
  const unsigned bs_off[2] = {(unsigned)(unsigned long long)&Bs[0][0],
                              (unsigned)(unsigned long long)&Bs[1][0]};
  const __bf16* arow = A + (size_t)m0 * K;
  const __bf16* brow = B + (size_t)n0 * K;
  if (wid == 0) {
    tdm_load_tile(arow, as_off[0], M, K);
    tdm_load_tile(brow, bs_off[0], N, K);
  }
#else
  const int ldr = tid >> 1;          // 0..127 tile row for LDS fill
  const int ldc = (tid & 1) << 4;    // 0 / 16
#endif

  for (int i = 0; i < nsteps; ++i) {
    const int cur = i & 1;
#if HAS_TDM
    __syncthreads();                 // previous compute done (LDS reuse safe)
    if (wid == 0) {
      if (i + 1 < nsteps) {
        const int nxt = cur ^ 1;
        tdm_load_tile(arow + (size_t)(i + 1) * BK, as_off[nxt], M, K);
        tdm_load_tile(brow + (size_t)(i + 1) * BK, bs_off[nxt], N, K);
        __builtin_amdgcn_s_wait_tensorcnt(2);   // in-order: tile i landed
      } else {
        __builtin_amdgcn_s_wait_tensorcnt(0);
      }
    }
    __syncthreads();                 // buffer `cur` ready for all waves
#else
    const int k0 = i * BK;
    const __bf16* ga = A + (size_t)(m0 + ldr) * K + (k0 + ldc);
    const __bf16* gb = B + (size_t)(n0 + ldr) * K + (k0 + ldc);
    const uint4 a0 = *(const uint4*)(ga);
    const uint4 a1 = *(const uint4*)(ga + 8);
    const uint4 b0 = *(const uint4*)(gb);
    const uint4 b1 = *(const uint4*)(gb + 8);
    __syncthreads();
    *(uint4*)(&As[cur][ldr * LDT + ldc])     = a0;
    *(uint4*)(&As[cur][ldr * LDT + ldc + 8]) = a1;
    *(uint4*)(&Bs[cur][ldr * LDT + ldc])     = b0;
    *(uint4*)(&Bs[cur][ldr * LDT + ldc + 8]) = b1;
    __syncthreads();
#endif

    const __bf16* abase = &As[cur][0];
    const __bf16* bbase = &Bs[cur][0];

    // A fragment (16x32): lane<16 holds K[0:8]+K[16:24], lane>=16 K[8:16]+K[24:32]
    v16bf af[2];
#pragma unroll
    for (int i2 = 0; i2 < 2; ++i2) {
      const __bf16* p = abase + (wm * 32 + i2 * 16 + fm) * LDT + kh * 8;
      union { v16bf v; uint4 q[2]; } u;
      u.q[0] = *(const uint4*)(p);
      u.q[1] = *(const uint4*)(p + 16);
      af[i2] = u.v;
    }
    // B fragment (32x16): lane n<16 holds col n K[0:8]+K[16:24]; lane>=16 K[8:16]+K[24:32]
    v16bf bfx[4];
#pragma unroll
    for (int j = 0; j < 4; ++j) {
      const __bf16* p = bbase + (wn * 64 + j * 16 + fm) * LDT + kh * 8;
      union { v16bf v; uint4 q[2]; } u;
      u.q[0] = *(const uint4*)(p);
      u.q[1] = *(const uint4*)(p + 16);
      bfx[j] = u.v;
    }
#pragma unroll
    for (int i2 = 0; i2 < 2; ++i2)
#pragma unroll
      for (int j = 0; j < 4; ++j)
        acc[i2][j] = __builtin_amdgcn_wmma_f32_16x16x32_bf16(
            false, af[i2], false, bfx[j], (short)0, acc[i2][j], false, false);
  }

  // C layout: VGPR r -> M = r + 8*(lane>>4); N = n-tile + (lane&15)
#pragma unroll
  for (int i = 0; i < 2; ++i)
#pragma unroll
    for (int j = 0; j < 4; ++j) {
      const int mt = m0 + wm * 32 + i * 16 + kh * 8;
      const int nn = n0 + wn * 64 + j * 16 + fm;
#pragma unroll
      for (int r = 0; r < 8; ++r) {
        const unsigned mm = (unsigned)(mt + r);
        const float val = acc[i][j][r];
        if (MODE == 0) {
          C[(size_t)(mm * (unsigned)N + (unsigned)nn)] = val;
        } else {
          // m = b*2048 + s ; n = h*64 + d ; dst[h][s][b][d]
          const unsigned bb = mm >> 11, ss = mm & 2047u;
          const unsigned hh = (unsigned)nn >> 6, dd = (unsigned)nn & 63u;
          C[(size_t)((hh << 19) + (ss << 8) + (bb << 6) + dd)] = val;
        }
      }
    }
}

// ---------------------------------------------------------------------------
// TTT scan: one workgroup per head, 128 threads (wave = batch row),
// lane owns elements (e, e+32). LN reductions are wave-local shuffles.
// Rank-4 weight update via V_WMMA_F32_16X16X4_F32 (K = 4 = batch).
// ---------------------------------------------------------------------------
#if __has_builtin(__builtin_amdgcn_wmma_f32_16x16x4_f32)
#define HAS_WMMA_F32K4 1
#else
#define HAS_WMMA_F32K4 0
#endif

__global__ __launch_bounds__(128)
void ttt_scan(const float* __restrict__ qh, const float* __restrict__ khp,
              const float* __restrict__ vhp,
              const float* __restrict__ fw_W, const float* __restrict__ fw_b,
              const float* __restrict__ fw_g, const float* __restrict__ fw_bl,
              const float* __restrict__ ttt_g, const float* __restrict__ ttt_b,
              float* __restrict__ attn)
{
  const int head = blockIdx.x;
  const int tid  = threadIdx.x;
  const int b    = tid >> 5;     // wave id == batch row
  const int lane = tid & 31;

  __shared__ float Wt[DHEAD][DHEAD + 1];     // Wt[e][d] = W[d][e], pad->stride 65
  __shared__ float bvec[DHEAD], gvec[DHEAD], blvec[DHEAD];
  __shared__ float kk[NBATCH][DHEAD + 4];    // current token k, stride 68
  __shared__ float dhs[NBATCH][DHEAD + 4];   // dL/dh
  __shared__ float colA[NBATCH][DHEAD];      // dpred * y
  __shared__ float colB[NBATCH][DHEAD];      // dpred

  for (int idx = tid; idx < DHEAD * DHEAD; idx += 128) {
    const int d = idx >> 6, e = idx & 63;
    Wt[e][d] = fw_W[((size_t)head * DHEAD + d) * DHEAD + e];
  }
  if (tid < DHEAD) {
    bvec[tid]  = fw_b[head * DHEAD + tid];
    gvec[tid]  = fw_g[head * DHEAD + tid];
    blvec[tid] = fw_bl[head * DHEAD + tid];
  }
  const float tg0 = ttt_g[head * DHEAD + lane];
  const float tg1 = ttt_g[head * DHEAD + lane + 32];
  const float tb0 = ttt_b[head * DHEAD + lane];
  const float tb1 = ttt_b[head * DHEAD + lane + 32];
  __syncthreads();

  const size_t hbase = (size_t)head * SEQ * (NBATCH * DHEAD);

  for (int s = 0; s < SEQ; ++s) {
    const size_t base = hbase + (size_t)s * (NBATCH * DHEAD) + b * DHEAD;
    const float k0 = khp[base + lane],  k1 = khp[base + lane + 32];
    const float w0 = vhp[base + lane],  w1 = vhp[base + lane + 32];
    const float q0 = qh[base + lane],   q1 = qh[base + lane + 32];
    kk[b][lane] = k0; kk[b][lane + 32] = k1;
    if (s + 1 < SEQ) {
      __builtin_prefetch(khp + base + NBATCH * DHEAD, 0, 3);
      __builtin_prefetch(vhp + base + NBATCH * DHEAD, 0, 3);
      __builtin_prefetch(qh  + base + NBATCH * DHEAD, 0, 3);
    }
    const float t0 = w0 - k0, t1 = w1 - k1;   // target = v - k

    for (int step = 0; step < 2; ++step) {
      // ---- forward: h = k @ W + b (k broadcast via shfl, W columns from LDS)
      float h0 = bvec[lane], h1 = bvec[lane + 32];
      for (int d = 0; d < 32; ++d) {
        const float ka = __shfl(k0, d, 32);
        const float kb = __shfl(k1, d, 32);
        h0 += ka * Wt[lane][d]      + kb * Wt[lane][d + 32];
        h1 += ka * Wt[lane + 32][d] + kb * Wt[lane + 32][d + 32];
      }
      // ---- LN forward + loss grad
      const float mu = wave_sum(h0 + h1) * (1.0f / 64.0f);
      const float sq = wave_sum(h0 * h0 + h1 * h1) * (1.0f / 64.0f);
      const float rs = rsqrtf(sq - mu * mu + EPSC);
      const float y0 = (h0 - mu) * rs, y1 = (h1 - mu) * rs;
      const float g0 = gvec[lane], g1 = gvec[lane + 32];
      const float r0 = y0 * g0 + blvec[lane]      - t0;
      const float r1 = y1 * g1 + blvec[lane + 32] - t1;
      const float cg = 2.0f / (float)(NBATCH * HEADS * DHEAD);
      const float dp0 = r0 * cg, dp1 = r1 * cg;
      // ---- LN backward
      const float dy0 = dp0 * g0, dy1 = dp1 * g1;
      const float m1 = wave_sum(dy0 + dy1) * (1.0f / 64.0f);
      const float m2 = wave_sum(dy0 * y0 + dy1 * y1) * (1.0f / 64.0f);
      const float dh0 = (dy0 - m1 - y0 * m2) * rs;
      const float dh1 = (dy1 - m1 - y1 * m2) * rs;
      dhs[b][lane] = dh0;       dhs[b][lane + 32] = dh1;
      colA[b][lane] = dp0 * y0; colA[b][lane + 32] = dp1 * y1;
      colB[b][lane] = dp0;      colB[b][lane + 32] = dp1;
      __syncthreads();

      // ---- vector-param grads (wave 0 sums over batch and applies)
      if (b == 0) {
#pragma unroll
        for (int hv = 0; hv < 2; ++hv) {
          const int e = lane + hv * 32;
          const float dg  = colA[0][e] + colA[1][e] + colA[2][e] + colA[3][e];
          const float dbl = colB[0][e] + colB[1][e] + colB[2][e] + colB[3][e];
          const float db  = dhs[0][e] + dhs[1][e] + dhs[2][e] + dhs[3][e];
          gvec[e]  -= LRC * dg;
          blvec[e] -= LRC * dbl;
          bvec[e]  -= LRC * db;
        }
      }

      // ---- rank-4 update: Wt[e][d] -= LR * sum_b dh[b][e] * k[b][d]
#if HAS_WMMA_F32K4
      {
        const int fm  = lane & 15;
        const int kh2 = lane >> 4;        // K-pair selector: b={0,1} / b={2,3}
#pragma unroll
        for (int j = 0; j < 4; ++j) {
          const int ti = (b << 2) | j;    // 16 tiles, 4 per wave
          const int e0 = (ti >> 2) << 4;
          const int d0 = (ti & 3) << 4;
          v2f afr, bfr;
          afr.x = -LRC * dhs[2 * kh2 + 0][e0 + fm];   // A[m=e][k=b]
          afr.y = -LRC * dhs[2 * kh2 + 1][e0 + fm];
          bfr.x = kk[2 * kh2 + 0][d0 + fm];           // B[k=b][n=d]
          bfr.y = kk[2 * kh2 + 1][d0 + fm];
          v8f cfr;
#pragma unroll
          for (int r = 0; r < 8; ++r)
            cfr[r] = Wt[e0 + r + 8 * kh2][d0 + fm];
          v8f dfr = __builtin_amdgcn_wmma_f32_16x16x4_f32(
              false, afr, false, bfr, (short)0, cfr, false, false);
#pragma unroll
          for (int r = 0; r < 8; ++r)
            Wt[e0 + r + 8 * kh2][d0 + fm] = dfr[r];
        }
      }
#else
      for (int idx = tid; idx < DHEAD * DHEAD; idx += 128) {
        const int e = idx >> 6, d = idx & 63;
        const float up = kk[0][d] * dhs[0][e] + kk[1][d] * dhs[1][e]
                       + kk[2][d] * dhs[2][e] + kk[3][d] * dhs[3][e];
        Wt[e][d] -= LRC * up;
      }
#endif
      __syncthreads();
    }

    // ---- final apply: z = LN(k@W+b)*g+bl; LN(z)*tg+tb; out = q + z
    float h0 = bvec[lane], h1 = bvec[lane + 32];
    for (int d = 0; d < 32; ++d) {
      const float ka = __shfl(k0, d, 32);
      const float kb = __shfl(k1, d, 32);
      h0 += ka * Wt[lane][d]      + kb * Wt[lane][d + 32];
      h1 += ka * Wt[lane + 32][d] + kb * Wt[lane + 32][d + 32];
    }
    const float mu = wave_sum(h0 + h1) * (1.0f / 64.0f);
    const float sq = wave_sum(h0 * h0 + h1 * h1) * (1.0f / 64.0f);
    const float rs = rsqrtf(sq - mu * mu + EPSC);
    const float z0 = ((h0 - mu) * rs) * gvec[lane]      + blvec[lane];
    const float z1 = ((h1 - mu) * rs) * gvec[lane + 32] + blvec[lane + 32];
    const float zmu = wave_sum(z0 + z1) * (1.0f / 64.0f);
    const float zsq = wave_sum(z0 * z0 + z1 * z1) * (1.0f / 64.0f);
    const float zrs = rsqrtf(zsq - zmu * zmu + EPSC);
    const float o0 = q0 + ((z0 - zmu) * zrs) * tg0 + tb0;
    const float o1 = q1 + ((z1 - zmu) * zrs) * tg1 + tb1;
    const size_t orow = ((size_t)b * SEQ + s) * EMB + head * DHEAD;
    attn[orow + lane]      = o0;
    attn[orow + lane + 32] = o1;
  }
}

// ---------------------------------------------------------------------------
// post-LN over E, * post_g + post_b, * gelu(gate_pre), cast to bf16
// ---------------------------------------------------------------------------
__global__ __launch_bounds__(256)
void post_gate(const float* __restrict__ attn, const float* __restrict__ gpre,
               const float* __restrict__ pg, const float* __restrict__ pb,
               __bf16* __restrict__ outg)
{
  __shared__ float red[16];
  const int row  = blockIdx.x;
  const int tid  = threadIdx.x;
  const int lane = tid & 31, wid = tid >> 5;
  const float* x = attn + (size_t)row * EMB;

  float v[4]; float s1 = 0.f, s2 = 0.f;
#pragma unroll
  for (int i = 0; i < 4; ++i) {
    const float t = x[tid + i * 256];
    v[i] = t; s1 += t; s2 += t * t;
  }
  s1 = wave_sum(s1); s2 = wave_sum(s2);
  if (lane == 0) { red[wid] = s1; red[wid + 8] = s2; }
  __syncthreads();
  float ts1 = 0.f, ts2 = 0.f;
#pragma unroll
  for (int ww = 0; ww < 8; ++ww) { ts1 += red[ww]; ts2 += red[ww + 8]; }
  const float mu = ts1 * (1.0f / (float)EMB);
  const float rs = rsqrtf(ts2 * (1.0f / (float)EMB) - mu * mu + EPSC);
#pragma unroll
  for (int i = 0; i < 4; ++i) {
    const int e = tid + i * 256;
    const float yn = (v[i] - mu) * rs * pg[e] + pb[e];
    const float gp = gpre[(size_t)row * EMB + e];
    const float gl = 0.5f * gp *
        (1.0f + tanhf(0.7978845608028654f * (gp + 0.044715f * gp * gp * gp)));
    outg[(size_t)row * EMB + e] = (__bf16)(yn * gl);
  }
}

// ---------------------------------------------------------------------------
extern "C" void kernel_launch(void* const* d_in, const int* in_sizes, int n_in,
                              void* d_out, int out_size, void* d_ws, size_t ws_size,
                              hipStream_t stream)
{
  (void)in_sizes; (void)n_in; (void)out_size; (void)ws_size;
  const float* x      = (const float*)d_in[0];
  const float* Wq     = (const float*)d_in[1];
  const float* Wk     = (const float*)d_in[2];
  const float* Wv     = (const float*)d_in[3];
  const float* Wo     = (const float*)d_in[4];
  const float* Wg     = (const float*)d_in[5];
  const float* fw_W   = (const float*)d_in[6];
  const float* fw_b   = (const float*)d_in[7];
  const float* fw_g   = (const float*)d_in[8];
  const float* fw_bl  = (const float*)d_in[9];
  const float* ttt_g  = (const float*)d_in[10];
  const float* ttt_b  = (const float*)d_in[11];
  const float* post_g = (const float*)d_in[12];
  const float* post_b = (const float*)d_in[13];

  char* w = (char*)d_ws;
  size_t off = 0;
  auto carve = [&](size_t bytes) -> char* {
    char* p = w + off;
    off = (off + bytes + 255) & ~(size_t)255;
    return p;
  };
  __bf16* xb     = (__bf16*)carve((size_t)TOK_M * EMB * 2);
  __bf16* wqb    = (__bf16*)carve((size_t)EMB * EMB * 2);
  __bf16* wkb    = (__bf16*)carve((size_t)EMB * EMB * 2);
  __bf16* wvb    = (__bf16*)carve((size_t)EMB * EMB * 2);
  __bf16* wgb    = (__bf16*)carve((size_t)EMB * EMB * 2);
  __bf16* wob    = (__bf16*)carve((size_t)EMB * EMB * 2);
  float*  qf     = (float*)carve((size_t)TOK_M * EMB * 4);   // [H,S,B,D]
  float*  kf     = (float*)carve((size_t)TOK_M * EMB * 4);
  float*  vf     = (float*)carve((size_t)TOK_M * EMB * 4);
  float*  gpre   = (float*)carve((size_t)TOK_M * EMB * 4);   // [B*S, E]
  float*  attn   = (float*)carve((size_t)TOK_M * EMB * 4);   // [B*S, E]
  __bf16* gatedb = (__bf16*)carve((size_t)TOK_M * EMB * 2);

  cast_f32_to_bf16<<<1024, 256, 0, stream>>>(x,  xb,  TOK_M * EMB);
  cast_f32_to_bf16<<<256, 256, 0, stream>>>(Wq, wqb, EMB * EMB);
  cast_f32_to_bf16<<<256, 256, 0, stream>>>(Wk, wkb, EMB * EMB);
  cast_f32_to_bf16<<<256, 256, 0, stream>>>(Wv, wvb, EMB * EMB);
  cast_f32_to_bf16<<<256, 256, 0, stream>>>(Wg, wgb, EMB * EMB);
  cast_f32_to_bf16<<<256, 256, 0, stream>>>(Wo, wob, EMB * EMB);

  dim3 gg(EMB / BN, TOK_M / BM);
  gemm_bf16_nt<1><<<gg, 256, 0, stream>>>(xb, wqb, qf,   TOK_M, EMB, EMB);
  gemm_bf16_nt<1><<<gg, 256, 0, stream>>>(xb, wkb, kf,   TOK_M, EMB, EMB);
  gemm_bf16_nt<1><<<gg, 256, 0, stream>>>(xb, wvb, vf,   TOK_M, EMB, EMB);
  gemm_bf16_nt<0><<<gg, 256, 0, stream>>>(xb, wgb, gpre, TOK_M, EMB, EMB);

  ttt_scan<<<HEADS, 128, 0, stream>>>(qf, kf, vf, fw_W, fw_b, fw_g, fw_bl,
                                      ttt_g, ttt_b, attn);

  post_gate<<<TOK_M, 256, 0, stream>>>(attn, gpre, post_g, post_b, gatedb);

  gemm_bf16_nt<0><<<gg, 256, 0, stream>>>(gatedb, wob, (float*)d_out,
                                          TOK_M, EMB, EMB);
}
